// SelfAttention_39651138076806
// MI455X (gfx1250) — matrix operand order repaired and checked by reading, not compiled, not run on hardware
//
#include <hip/hip_runtime.h>

typedef __attribute__((ext_vector_type(16))) _Float16 v16h;
typedef __attribute__((ext_vector_type(8)))  float    v8f;

#define BATCH 4
#define SEQ   4096
#define DIN   256
#define DATT  128
#define BS    (BATCH*SEQ)

union V16H { v16h v; uint4 u[2]; };
union U4H8 { uint4 q; _Float16 h[8]; };
union H4   { _Float16 h[4]; uint2 u2; };

// gfx1250 async copy: global -> LDS, 16B per lane, tracked by ASYNCcnt
__device__ __forceinline__ void async_ld_b128(unsigned lds_off, const void* gaddr) {
    asm volatile("global_load_async_to_lds_b128 %0, %1, off"
                 :: "v"(lds_off), "v"(gaddr) : "memory");
}
__device__ __forceinline__ void wait_async0() {
    asm volatile("s_wait_asynccnt 0x0" ::: "memory");
}

// ---------------------------------------------------------------------------
// Kernel 1: out[row, n] = f16( x[row,:] @ W[:,n] + bias[n] )
// Block: 256 threads (8 waves). Each wave: 16 rows x 128 cols.
// ---------------------------------------------------------------------------
__global__ void __launch_bounds__(256)
qkv_proj_kernel(const float* __restrict__ x,
                const float* __restrict__ W,
                const float* __restrict__ bias,
                _Float16* __restrict__ out)
{
    __shared__ __align__(16) _Float16 xlds[128 * 32];   // [row][k]
    __shared__ __align__(16) _Float16 wldsT[128 * 32];  // [n][k]

    const int tid  = threadIdx.x;
    const int lane = tid & 31;
    const int wave = tid >> 5;
    const int h    = lane >> 4;
    const int ln   = lane & 15;
    const int row0 = blockIdx.x * 128;

    v8f zero = {};
    v8f acc[8];
#pragma unroll
    for (int n = 0; n < 8; ++n) acc[n] = zero;

    for (int kc = 0; kc < 8; ++kc) {
        __syncthreads();
        // stage x chunk: 1024 float4 loads, packed f16 b64 stores
#pragma unroll
        for (int i = 0; i < 4; ++i) {
            int u = tid + i * 256;                       // 0..1023
            const float* xr = x + (size_t)(row0 + (u >> 3)) * DIN + kc * 32 + (u & 7) * 4;
            float4 f = *(const float4*)xr;
            H4 p;
            p.h[0] = (_Float16)f.x; p.h[1] = (_Float16)f.y;
            p.h[2] = (_Float16)f.z; p.h[3] = (_Float16)f.w;
            ((uint2*)xlds)[u] = p.u2;
        }
        // stage W chunk transposed
#pragma unroll
        for (int i = 0; i < 4; ++i) {
            int u = tid + i * 256;
            int kk = u >> 5, nn = (u & 31) * 4;
            const float* wr = W + (size_t)(kc * 32 + kk) * DATT + nn;
            float4 f = *(const float4*)wr;
            H4 p;
            p.h[0] = (_Float16)f.x; p.h[1] = (_Float16)f.y;
            p.h[2] = (_Float16)f.z; p.h[3] = (_Float16)f.w;
#pragma unroll
            for (int jj = 0; jj < 4; ++jj)
                wldsT[(nn + jj) * 32 + kk] = p.h[jj];
        }
        __syncthreads();

        // A fragment (16x32): lane row = ln; K runs h*8..+7 and 16+h*8..+7
        V16H a;
        const uint4* xrow = (const uint4*)&xlds[(wave * 16 + ln) * 32];
        a.u[0] = xrow[h];
        a.u[1] = xrow[2 + h];

        // preload all 8 B fragments, then 8 back-to-back WMMAs
        V16H bf[8];
#pragma unroll
        for (int n = 0; n < 8; ++n) {
            const uint4* wrow = (const uint4*)&wldsT[(n * 16 + ln) * 32];
            bf[n].u[0] = wrow[h];
            bf[n].u[1] = wrow[2 + h];
        }
#pragma unroll
        for (int n = 0; n < 8; ++n)
            acc[n] = __builtin_amdgcn_wmma_f32_16x16x32_f16(
                false, a.v, false, bf[n].v, (short)0, acc[n], false, false);
    }

#pragma unroll
    for (int n = 0; n < 8; ++n) {
        float bb = bias[n * 16 + ln];
#pragma unroll
        for (int r = 0; r < 8; ++r) {
            int row = row0 + wave * 16 + h * 8 + r;
            out[(size_t)row * DATT + n * 16 + ln] = (_Float16)(acc[n][r] + bb);
        }
    }
}

// ---------------------------------------------------------------------------
// Kernel 2: flash attention, double-buffered LDS, async K staging.
// Each wave owns 16 query rows; block = 128 rows; key chunks of 64.
// ---------------------------------------------------------------------------
__global__ void __launch_bounds__(256)
attn_kernel(const _Float16* __restrict__ q,
            const _Float16* __restrict__ k,
            const _Float16* __restrict__ v,
            float* __restrict__ out)
{
    __shared__ __align__(16) _Float16 klds[2][64 * 128];    // [kv][d]
    __shared__ __align__(16) _Float16 vldsT[2][128 * 64];   // [d][kv]
    __shared__ __align__(16) _Float16 plds[8][16 * 64];     // per-wave P tile

    const int tid  = threadIdx.x;
    const int lane = tid & 31;
    const int wave = tid >> 5;
    const int h    = lane >> 4;
    const int ln   = lane & 15;
    const int b    = blockIdx.x >> 5;
    const int q0   = (blockIdx.x & 31) * 128;
    const float scale = 0.08838834764831845f;   // 1/sqrt(128)
    const int NC = SEQ / 64;

    // Q fragments (A layout: lane row = ln, both halves)
    const int qrow = b * SEQ + q0 + wave * 16 + ln;
    const uint4* qrowp = (const uint4*)&q[(size_t)qrow * DATT];
    V16H qa[4];
#pragma unroll
    for (int dc = 0; dc < 4; ++dc) {
        qa[dc].u[0] = qrowp[dc * 4 + h];
        qa[dc].u[1] = qrowp[dc * 4 + 2 + h];
    }

    v8f zero = {};
    v8f acc[8];
#pragma unroll
    for (int i = 0; i < 8; ++i) acc[i] = zero;
    float m_run[8], l_run[8];
#pragma unroll
    for (int r = 0; r < 8; ++r) { m_run[r] = -3.0e38f; l_run[r] = 0.0f; }

    _Float16* pw = plds[wave];
    uint4 vreg[4];

    auto issueK = [&](int buf, int c) {
        const _Float16* kg = &k[((size_t)b * SEQ + c * 64) * DATT];
#pragma unroll
        for (int i = 0; i < 4; ++i) {
            int u = tid + i * 256;
            unsigned loff = (unsigned)(unsigned long long)(void*)&klds[buf][u * 8];
            async_ld_b128(loff, kg + (size_t)u * 8);
        }
    };
    auto loadV = [&](int c) {
        const uint4* vg = (const uint4*)&v[((size_t)b * SEQ + c * 64) * DATT];
#pragma unroll
        for (int i = 0; i < 4; ++i) vreg[i] = vg[tid + i * 256];
    };
    auto scatterV = [&](int buf) {
#pragma unroll
        for (int i = 0; i < 4; ++i) {
            U4H8 t; t.q = vreg[i];
            int u = tid + i * 256;
            int kv = u >> 4, dbase = (u & 15) * 8;
#pragma unroll
            for (int jj = 0; jj < 8; ++jj)
                vldsT[buf][(dbase + jj) * 64 + kv] = t.h[jj];
        }
    };

    // prologue: chunk 0 into buffer 0
    issueK(0, 0);
    loadV(0);
    scatterV(0);
    wait_async0();
    __syncthreads();

    for (int c = 0; c < NC; ++c) {
        const int p = c & 1;
        const bool have_next = (c + 1) < NC;
        if (have_next) {            // stream next chunk while we compute
            issueK(1 - p, c + 1);
            loadV(c + 1);
        }

        // ---- S = Q @ K^T : 4 tiles, batched fragment loads ----
        v8f s[4];
#pragma unroll
        for (int n = 0; n < 4; ++n) {
            V16H kf[4];
            const uint4* krow = (const uint4*)&klds[p][(n * 16 + ln) * 128];
#pragma unroll
            for (int dc = 0; dc < 4; ++dc) {
                kf[dc].u[0] = krow[dc * 4 + h];
                kf[dc].u[1] = krow[dc * 4 + 2 + h];
            }
            v8f cacc = zero;
#pragma unroll
            for (int dc = 0; dc < 4; ++dc)
                cacc = __builtin_amdgcn_wmma_f32_16x16x32_f16(
                    false, qa[dc].v, false, kf[dc].v, (short)0, cacc, false, false);
            s[n] = cacc;
        }

        // ---- online softmax ----
#pragma unroll
        for (int n = 0; n < 4; ++n)
#pragma unroll
            for (int r = 0; r < 8; ++r) s[n][r] *= scale;

#pragma unroll
        for (int r = 0; r < 8; ++r) {
            float mx = fmaxf(fmaxf(s[0][r], s[1][r]), fmaxf(s[2][r], s[3][r]));
#pragma unroll
            for (int off = 1; off < 16; off <<= 1)
                mx = fmaxf(mx, __shfl_xor(mx, off, 32));
            float m_new = fmaxf(m_run[r], mx);
            float alpha = __expf(m_run[r] - m_new);
            float rs = 0.0f;
#pragma unroll
            for (int n = 0; n < 4; ++n) {
                float pv = __expf(s[n][r] - m_new);
                s[n][r] = pv;
                rs += pv;
            }
#pragma unroll
            for (int off = 1; off < 16; off <<= 1)
                rs += __shfl_xor(rs, off, 32);
            l_run[r] = l_run[r] * alpha + rs;
            m_run[r] = m_new;
#pragma unroll
            for (int dt = 0; dt < 8; ++dt) acc[dt][r] *= alpha;
        }

        // V for next chunk: scatter-transpose into the other buffer
        // (global loads above have had the S-phase to complete)
        if (have_next) scatterV(1 - p);

        // ---- P: C-layout -> A-layout via per-wave LDS tile ----
#pragma unroll
        for (int n = 0; n < 4; ++n)
#pragma unroll
            for (int r = 0; r < 8; ++r)
                pw[(h * 8 + r) * 64 + n * 16 + ln] = (_Float16)s[n][r];

        // ---- O += P @ V ----
#pragma unroll
        for (int kvc = 0; kvc < 2; ++kvc) {
            V16H pa;
            const uint4* prow = (const uint4*)&pw[ln * 64];
            pa.u[0] = prow[kvc * 4 + h];
            pa.u[1] = prow[kvc * 4 + 2 + h];
#pragma unroll
            for (int g = 0; g < 2; ++g) {
                V16H vf[4];
#pragma unroll
                for (int t = 0; t < 4; ++t) {
                    const uint4* vrow = (const uint4*)&vldsT[p][((g * 4 + t) * 16 + ln) * 64];
                    vf[t].u[0] = vrow[kvc * 4 + h];
                    vf[t].u[1] = vrow[kvc * 4 + 2 + h];
                }
#pragma unroll
                for (int t = 0; t < 4; ++t)
                    acc[g * 4 + t] = __builtin_amdgcn_wmma_f32_16x16x32_f16(
                        false, pa.v, false, vf[t].v, (short)0, acc[g * 4 + t], false, false);
            }
        }

        if (have_next) wait_async0();   // next chunk's K copy must be in LDS
        __syncthreads();                // all waves done reading buffers p
    }

    // ---- finalize ----
#pragma unroll
    for (int dt = 0; dt < 8; ++dt) {
#pragma unroll
        for (int r = 0; r < 8; ++r) {
            int row = b * SEQ + q0 + wave * 16 + h * 8 + r;
            out[(size_t)row * DATT + dt * 16 + ln] = acc[dt][r] / l_run[r];
        }
    }
}

// ---------------------------------------------------------------------------
extern "C" void kernel_launch(void* const* d_in, const int* in_sizes, int n_in,
                              void* d_out, int out_size, void* d_ws, size_t ws_size,
                              hipStream_t stream)
{
    (void)in_sizes; (void)n_in; (void)out_size; (void)ws_size;
    const float* x  = (const float*)d_in[0];
    const float* Wq = (const float*)d_in[1];
    const float* bq = (const float*)d_in[2];
    const float* Wk = (const float*)d_in[3];
    const float* bk = (const float*)d_in[4];
    const float* Wv = (const float*)d_in[5];
    const float* bv = (const float*)d_in[6];
    float* out = (float*)d_out;

    _Float16* qh = (_Float16*)d_ws;
    _Float16* kh = qh + (size_t)BS * DATT;
    _Float16* vh = kh + (size_t)BS * DATT;

    dim3 blk(256);
    dim3 grd(BS / 128);
    qkv_proj_kernel<<<grd, blk, 0, stream>>>(x, Wq, bq, qh);
    qkv_proj_kernel<<<grd, blk, 0, stream>>>(x, Wk, bk, kh);
    qkv_proj_kernel<<<grd, blk, 0, stream>>>(x, Wv, bv, vh);
    attn_kernel<<<grd, blk, 0, stream>>>(qh, kh, vh, out);
}
